// controversialnet_12197707120649
// MI455X (gfx1250) — compile-verified, matrix operand order checked by reading, not run on hardware
//
#include <hip/hip_runtime.h>
#include <hip/hip_bf16.h>
#include <math.h>

#define NN 8192
#define DIN 256
#define KC 8

typedef __attribute__((ext_vector_type(16))) _Float16 v16h;
typedef __attribute__((ext_vector_type(2)))  __fp16   v2fp16;
typedef __attribute__((ext_vector_type(8)))  float    v8f;

// ---------------------------------------------------------------------------
// f16 packing: single v_cvt_pk_rtz_f16_f32 per float pair (no sub-word
// reassembly -- the builtin's result is already the packed 32-bit value).
// ---------------------------------------------------------------------------
__device__ __forceinline__ unsigned pack2(float x, float y) {
  v2fp16 h = __builtin_amdgcn_cvt_pkrtz(x, y);
  union { v2fp16 h; unsigned u; } c;
  c.h = h;
  return c.u;
}
__device__ __forceinline__ uint2 pack4(float4 f) {
  uint2 u;
  u.x = pack2(f.x, f.y);
  u.y = pack2(f.z, f.w);
  return u;
}

// ---------------------------------------------------------------------------
// WMMA 16x16x32 f16 fragment load from an LDS tile.
// Per ISA 7.12.2 (16-bit A 16x32): lane L holds row m = L&15; half = L>>4
// selects K pairs {half*8..+7} and {16+half*8..+7} -> two ds_load_b128 per
// fragment for a row-major LDS tile (stride in halves, 16B aligned).
// B tiles are stored column-major ([n][k]) so the same loader serves both.
// ---------------------------------------------------------------------------
__device__ __forceinline__ v16h load_frag(const _Float16* base, int stride_h, int lane) {
  int m = lane & 15;
  int h = lane >> 4;
  const uint4* row = (const uint4*)(base + m * stride_h);
  union { v16h v; uint4 q[2]; } u;
  u.q[0] = row[h];
  u.q[1] = row[2 + h];
  return u.v;
}

#define WMMA_F32_F16(a, b, c) \
  __builtin_amdgcn_wmma_f32_16x16x32_f16(false, (a), false, (b), (short)0, (c), false, false)

// ---------------------------------------------------------------------------
// Generic C[M,N] = act(A[M,K] @ B[K,N] + bias), fp32 in/out, f16 WMMA math.
// Block tile 128(M) x 64(N), 256 threads = 8 waves.
// Software-pipelined: global b128 fetch of tile t+1 is issued before the WMMA
// compute of tile t; the loadcnt wait + cvt + ds_store happen after compute.
// grid = (N/64, M/128); K multiple of 32. act: 0 = none, 1 = tanh
// ---------------------------------------------------------------------------
__global__ __launch_bounds__(256)
void gemm_wmma_kernel(const float* __restrict__ A, const float* __restrict__ B,
                      const float* __restrict__ bias, float* __restrict__ C,
                      int M, int N, int K, int act) {
  __shared__ __attribute__((aligned(16))) _Float16 As[128 * 32]; // row-major [m][k]
  __shared__ __attribute__((aligned(16))) _Float16 Bs[64 * 32];  // col-major [n][k]

  const int m0 = blockIdx.y * 128;
  const int n0 = blockIdx.x * 64;
  const int t = threadIdx.x;
  const int wave = t >> 5;
  const int lane = t & 31;

  // A staging map: thread t -> rows {rt, rt+32, rt+64, rt+96}, cols ca..ca+3
  const int rt = t >> 3;
  const int ca = (t & 7) * 4;
  const float* aBase = A + (size_t)(m0 + rt) * K + ca;
  const size_t aRowStep = (size_t)32 * K;
  // B staging map: thread t -> rows {2p, 2p+1}, cols nb..nb+3 (transposed into LDS)
  const int p = t >> 4;
  const int nb = (t & 15) * 4;
  const float* bBase = B + (size_t)(2 * p) * N + n0 + nb;

  float4 aR[4];
  float4 bR[2];

  auto fetch = [&](int k0) {
    #pragma unroll
    for (int i = 0; i < 4; ++i)
      aR[i] = *(const float4*)(aBase + aRowStep * i + k0);
    const float* bp = bBase + (size_t)k0 * N;
    bR[0] = *(const float4*)(bp);
    bR[1] = *(const float4*)(bp + N);
  };
  auto stage = [&]() {
    #pragma unroll
    for (int i = 0; i < 4; ++i)
      *(uint2*)&As[(rt + i * 32) * 32 + ca] = pack4(aR[i]);
    const float* b0 = (const float*)&bR[0];
    const float* b1 = (const float*)&bR[1];
    #pragma unroll
    for (int q = 0; q < 4; ++q)
      *(unsigned*)&Bs[(nb + q) * 32 + 2 * p] = pack2(b0[q], b1[q]);
  };

  v8f acc[4] = {};

  fetch(0);
  stage();
  for (int k0 = 0; k0 < K; k0 += 32) {
    __syncthreads();                    // staged tile visible to all waves
    const bool hasNext = (k0 + 32) < K;
    if (hasNext) {
      fetch(k0 + 32);                   // issue global b128 loads early
      if (k0 + 64 < K) {                // prefetch tile after next (global_prefetch_b8)
        __builtin_prefetch(aBase + k0 + 64, 0, 0);
        __builtin_prefetch(bBase + (size_t)(k0 + 64) * N, 0, 0);
      }
    }
    // WMMA compute on current tile (overlaps in-flight global loads)
    v16h a = load_frag(As + wave * 16 * 32, 32, lane);
    #pragma unroll
    for (int s = 0; s < 4; ++s) {
      v16h b = load_frag(Bs + s * 16 * 32, 32, lane);
      acc[s] = WMMA_F32_F16(a, b, acc[s]);
    }
    __syncthreads();                    // all waves done reading the tile
    if (hasNext) stage();               // wait loadcnt, cvt_pk, ds_store
  }

  // Epilogue: C/D layout: lane n = L&15, VGPR r -> row r + 8*(L>>4)
  const int n = lane & 15;
  const int h = lane >> 4;
  #pragma unroll
  for (int s = 0; s < 4; ++s) {
    int col = n0 + s * 16 + n;
    float bv = bias ? bias[col] : 0.0f;
    #pragma unroll
    for (int r = 0; r < 8; ++r) {
      int row = m0 + wave * 16 + r + 8 * h;
      float v = acc[s][r] + bv;
      if (act == 1) v = tanhf(v);
      C[(size_t)row * N + col] = v;
    }
  }
}

// ---------------------------------------------------------------------------
// n2[i] = sum_c E[i][c]^2   (E = new_features [8192,64], fp32, float4 reads)
// ---------------------------------------------------------------------------
__global__ __launch_bounds__(256)
void rownorm_kernel(const float* __restrict__ E, float* __restrict__ n2) {
  int i = blockIdx.x * blockDim.x + threadIdx.x;
  if (i < NN) {
    const float4* row = (const float4*)(E + (size_t)i * 64);
    float s = 0.0f;
    #pragma unroll
    for (int c = 0; c < 16; ++c) {
      float4 v = row[c];
      s += v.x * v.x + v.y * v.y + v.z * v.z + v.w * v.w;
    }
    n2[i] = s;
  }
}

// ---------------------------------------------------------------------------
// new_force[i][j] = sqrt(max(n2[i]+n2[j]-2*(E Et)[i][j], 0)) * path_forces[i][j]
// 128x128 tile per block via WMMA gram (K = 64 -> two k-steps of 32).
// grid = (64, 64). b128 staging; fused epilogue writes force exactly once.
// ---------------------------------------------------------------------------
__global__ __launch_bounds__(256)
void force_kernel(const float* __restrict__ E, const float* __restrict__ n2,
                  const float* __restrict__ path_forces, float* __restrict__ force) {
  __shared__ __attribute__((aligned(16))) _Float16 Ei[128 * 64]; // A: rows i, [m][k]
  __shared__ __attribute__((aligned(16))) _Float16 Ej[128 * 64]; // B^T == Ej row-major [n][k]
  __shared__ float n2i[128];
  __shared__ float n2j[128];

  const int i0 = blockIdx.y * 128;
  const int j0 = blockIdx.x * 128;
  const int t = threadIdx.x;
  const int wave = t >> 5;
  const int lane = t & 31;

  // staging: granule g = t + i*256 -> row g>>4, cols (g&15)*4 (float4 each)
  #pragma unroll
  for (int i = 0; i < 8; ++i) {
    int g = t + i * 256;
    int r = g >> 4;
    int c = (g & 15) * 4;
    float4 vi = *(const float4*)(E + (size_t)(i0 + r) * 64 + c);
    float4 vj = *(const float4*)(E + (size_t)(j0 + r) * 64 + c);
    *(uint2*)&Ei[r * 64 + c] = pack4(vi);
    *(uint2*)&Ej[r * 64 + c] = pack4(vj);
  }
  if (t < 128) n2i[t] = n2[i0 + t];
  else         n2j[t - 128] = n2[j0 + (t - 128)];
  __syncthreads();

  v8f acc[8] = {};
  #pragma unroll
  for (int ks = 0; ks < 2; ++ks) {
    v16h a = load_frag(Ei + wave * 16 * 64 + ks * 32, 64, lane);
    #pragma unroll
    for (int s = 0; s < 8; ++s) {
      v16h b = load_frag(Ej + s * 16 * 64 + ks * 32, 64, lane);
      acc[s] = WMMA_F32_F16(a, b, acc[s]);
    }
  }

  const int n = lane & 15;
  const int h = lane >> 4;
  #pragma unroll
  for (int s = 0; s < 8; ++s) {
    int j = j0 + s * 16 + n;
    float nj = n2j[s * 16 + n];
    #pragma unroll
    for (int r = 0; r < 8; ++r) {
      int li = wave * 16 + r + 8 * h;
      int i = i0 + li;
      float sq = n2i[li] + nj - 2.0f * acc[s][r];
      float d = (sq > 0.0f) ? sqrtf(sq) : 0.0f;
      size_t off = (size_t)i * NN + j;
      force[off] = d * path_forces[off];
    }
  }
}

// ---------------------------------------------------------------------------
// Fg[i][k] = sum_j force[i][j] * members[j][k]  (N=8 tail -> VALU reduction)
// one block per row i; float4 force reads, LDS tree reduce.
// ---------------------------------------------------------------------------
__global__ __launch_bounds__(256)
void fg_kernel(const float* __restrict__ force, const float* __restrict__ members,
               float* __restrict__ Fg) {
  __shared__ float red[256 * KC];
  const int i = blockIdx.x;
  const int t = threadIdx.x;
  float acc[KC] = {};
  for (int c = t; c < NN / 4; c += 256) {
    int j = c * 4;
    float4 f = *(const float4*)(force + (size_t)i * NN + j);
    const float* m = members + (size_t)j * KC;
    #pragma unroll
    for (int k = 0; k < KC; ++k)
      acc[k] += f.x * m[k] + f.y * m[k + 8] + f.z * m[k + 16] + f.w * m[k + 24];
  }
  #pragma unroll
  for (int k = 0; k < KC; ++k) red[t * KC + k] = acc[k];
  __syncthreads();
  for (int s = 128; s > 0; s >>= 1) {
    if (t < s)
      #pragma unroll
      for (int k = 0; k < KC; ++k) red[t * KC + k] += red[(t + s) * KC + k];
    __syncthreads();
  }
  if (t < KC) Fg[i * KC + t] = red[t];
}

__global__ void zero_kernel(float* p, int n) {
  int i = blockIdx.x * blockDim.x + threadIdx.x;
  if (i < n) p[i] = 0.0f;
}

// ---------------------------------------------------------------------------
// d0raw[c] += sum_{r in chunk} Fg[r] * W_d0[r][c]  (GEMV over 256 MB weight)
// grid = (1, 512): each block covers all 1024 cols (float4/thread) over a
// 128-row chunk; float4 weight reads keep HBM at full rate; atomic combine.
// ---------------------------------------------------------------------------
__global__ __launch_bounds__(256)
void d0_kernel(const float* __restrict__ Fg, const float* __restrict__ W_d0,
               float* __restrict__ d0raw) {
  const int c = threadIdx.x * 4;          // 0..1020
  const int r0 = blockIdx.y * 128;
  float4 acc = make_float4(0.f, 0.f, 0.f, 0.f);
  for (int r = r0; r < r0 + 128; ++r) {
    float g = Fg[r];
    float4 w = *(const float4*)(W_d0 + (size_t)r * 1024 + c);
    acc.x += g * w.x; acc.y += g * w.y; acc.z += g * w.z; acc.w += g * w.w;
  }
  atomicAdd(&d0raw[c + 0], acc.x);
  atomicAdd(&d0raw[c + 1], acc.y);
  atomicAdd(&d0raw[c + 2], acc.z);
  atomicAdd(&d0raw[c + 3], acc.w);
}

// ---------------------------------------------------------------------------
// Head: relu(d0raw+b_d0) -> relu(@W_d1+b_d1) -> @W_d2 -> softmax(8). 1 block.
// ---------------------------------------------------------------------------
__global__ __launch_bounds__(256)
void head_kernel(const float* __restrict__ d0raw, const float* __restrict__ b_d0,
                 const float* __restrict__ W_d1, const float* __restrict__ b_d1,
                 const float* __restrict__ W_d2, float* __restrict__ out) {
  __shared__ float a0[1024];
  __shared__ float a1[256];
  __shared__ float logits[KC];
  const int t = threadIdx.x;
  for (int i = t; i < 1024; i += 256) {
    float v = d0raw[i] + b_d0[i];
    a0[i] = v > 0.0f ? v : 0.0f;
  }
  __syncthreads();
  float acc = b_d1[t];
  for (int i = 0; i < 1024; ++i) acc += a0[i] * W_d1[(size_t)i * 256 + t];
  a1[t] = acc > 0.0f ? acc : 0.0f;
  __syncthreads();
  if (t < KC) {
    float l = 0.0f;
    for (int i = 0; i < 256; ++i) l += a1[i] * W_d2[(size_t)i * KC + t];
    logits[t] = l;
  }
  __syncthreads();
  if (t == 0) {
    float mx = logits[0];
    for (int k = 1; k < KC; ++k) mx = fmaxf(mx, logits[k]);
    float s = 0.0f, e[KC];
    for (int k = 0; k < KC; ++k) { e[k] = expf(logits[k] - mx); s += e[k]; }
    for (int k = 0; k < KC; ++k) out[k] = e[k] / s;
  }
}

// ---------------------------------------------------------------------------
// Launch.  Inputs (setup_inputs order, all fp32):
//  0 F_ [N,N]  1 X [N,256]  2 W_e0 3 b_e0 4 W_e1 5 b_e1 6 W_e2 7 b_e2
//  8 W_d0 [65536,1024] 9 b_d0 10 W_d1 [1024,256] 11 b_d1 12 W_d2 [256,8]
// 13 path_forces [N,N] 14 members [N,8]
// d_out: new_features (524288) | new_force (67108864) | output (8)
// ---------------------------------------------------------------------------
extern "C" void kernel_launch(void* const* d_in, const int* in_sizes, int n_in,
                              void* d_out, int out_size, void* d_ws, size_t ws_size,
                              hipStream_t stream) {
  const float* F_   = (const float*)d_in[0];
  const float* X    = (const float*)d_in[1];
  const float* W_e0 = (const float*)d_in[2];
  const float* b_e0 = (const float*)d_in[3];
  const float* W_e1 = (const float*)d_in[4];
  const float* b_e1 = (const float*)d_in[5];
  const float* W_e2 = (const float*)d_in[6];
  const float* b_e2 = (const float*)d_in[7];
  const float* W_d0 = (const float*)d_in[8];
  const float* b_d0 = (const float*)d_in[9];
  const float* W_d1 = (const float*)d_in[10];
  const float* b_d1 = (const float*)d_in[11];
  const float* W_d2 = (const float*)d_in[12];
  const float* path_forces = (const float*)d_in[13];
  const float* members     = (const float*)d_in[14];

  float* out_nf    = (float*)d_out;                       // [8192,64]
  float* out_force = out_nf + (size_t)NN * 64;            // [8192,8192]
  float* out_soft  = out_force + (size_t)NN * NN;         // [8]

  // workspace (fp32): buf0 / buf1 ping-pong (8 MB each), then small tails
  float* buf0  = (float*)d_ws;                            // inp, later h2
  float* buf1  = buf0 + (size_t)NN * DIN;                 // h1
  float* n2    = buf1 + (size_t)NN * DIN;                 // [8192]
  float* Fg    = n2 + NN;                                 // [65536]
  float* d0raw = Fg + (size_t)NN * KC;                    // [1024]

  dim3 blk(256);

  // 1) inp = F_ @ X                        [8192,8192]x[8192,256]
  gemm_wmma_kernel<<<dim3(DIN / 64, NN / 128), blk, 0, stream>>>(
      F_, X, nullptr, buf0, NN, DIN, NN, 0);
  // 2) h1 = tanh(inp @ W_e0 + b_e0)
  gemm_wmma_kernel<<<dim3(256 / 64, NN / 128), blk, 0, stream>>>(
      buf0, W_e0, b_e0, buf1, NN, 256, 256, 1);
  // 3) h2 = tanh(h1 @ W_e1 + b_e1)
  gemm_wmma_kernel<<<dim3(128 / 64, NN / 128), blk, 0, stream>>>(
      buf1, W_e1, b_e1, buf0, NN, 128, 256, 1);
  // 4) new_features = h2 @ W_e2 + b_e2  -> d_out
  gemm_wmma_kernel<<<dim3(64 / 64, NN / 128), blk, 0, stream>>>(
      buf0, W_e2, b_e2, out_nf, NN, 64, 128, 0);
  // 5) row norms
  rownorm_kernel<<<NN / 256, blk, 0, stream>>>(out_nf, n2);
  // 6) new_force = cdist * path_forces (fused WMMA gram + epilogue)
  force_kernel<<<dim3(NN / 128, NN / 128), blk, 0, stream>>>(
      out_nf, n2, path_forces, out_force);
  // 7) Fg = new_force @ members
  fg_kernel<<<NN, blk, 0, stream>>>(out_force, members, Fg);
  // 8) discriminator layer 0 (GEMV over 256 MB weight)
  zero_kernel<<<(1024 + 255) / 256, blk, 0, stream>>>(d0raw, 1024);
  d0_kernel<<<dim3(1, 512), blk, 0, stream>>>(Fg, W_d0, d0raw);
  // 9) head: relu -> relu -> softmax
  head_kernel<<<1, blk, 0, stream>>>(d0raw, b_d0, W_d1, b_d1, W_d2, out_soft);
}
